// _HFAttention_5162550689804
// MI455X (gfx1250) — matrix operand order repaired and checked by reading, not compiled, not run on hardware
//
#include <hip/hip_runtime.h>

typedef unsigned short u16;
typedef unsigned int   u32;
typedef __attribute__((ext_vector_type(4)))  u32    u32x4;
typedef __attribute__((ext_vector_type(2)))  u32    u32x2;
typedef __attribute__((ext_vector_type(16))) __bf16 v16bf;
typedef __attribute__((ext_vector_type(8)))  float  v8f;

constexpr int S_  = 4096;
constexpr int H_  = 2048;
constexpr int NH_ = 16;
constexpr int NKV_= 4;
constexpr int D_  = 128;

union FragU { v16bf v; u32x4 q[2]; };

__device__ __forceinline__ u16 f2bf(float x) {
  union { float f; u32 u; } v; v.f = x;
  u32 r = v.u + 0x7fffu + ((v.u >> 16) & 1u);
  return (u16)(r >> 16);
}

__device__ __forceinline__ v8f vzero() {
  v8f z;
#pragma unroll
  for (int i = 0; i < 8; ++i) z[i] = 0.0f;
  return z;
}

// A-matrix 16x32 bf16 fragment from row-major (ld elements), base at tile (0,0).
// lane<16: row=lane, K = {0..7, 16..23}; lane>=16: row=lane-16, K = {8..15, 24..31}
__device__ __forceinline__ v16bf load_frag_a(const u16* p, int ld, int lane) {
  int row = lane & 15;
  int kh  = (lane >> 4) << 3;           // 0 or 8
  FragU f;
  f.q[0] = *(const u32x4*)(p + (size_t)row * ld + kh);
  f.q[1] = *(const u32x4*)(p + (size_t)row * ld + kh + 16);
  return f.v;
}

// B-matrix 32x16 bf16 fragment, loaded from B^T stored row-major (row = n, ld elements).
// lane<16: col n=lane, K = {0..7, 16..23}; lane>=16: col n=lane-16, K = {8..15, 24..31}
__device__ __forceinline__ v16bf load_frag_b(const u16* p, int ld, int lane) {
  int row = lane & 15;
  int k0  = (lane >> 4) << 3;           // 0 or 8
  FragU f;
  f.q[0] = *(const u32x4*)(p + (size_t)row * ld + k0);
  f.q[1] = *(const u32x4*)(p + (size_t)row * ld + k0 + 16);
  return f.v;
}

__device__ __forceinline__ v8f wmma_bf16(v16bf a, v16bf b, v8f c) {
  return __builtin_amdgcn_wmma_f32_16x16x32_bf16(false, a, false, b, (short)0, c, false, false);
}

// ---------------- f32 -> bf16 conversion ----------------
__global__ __launch_bounds__(256) void cvt_kernel(const float* __restrict__ in,
                                                  u16* __restrict__ out, int n) {
  int i = (blockIdx.x * 256 + threadIdx.x) * 4;
  if (i >= n) return;
  float4 v = *(const float4*)(in + i);
  u32x2 o;
  o[0] = (u32)f2bf(v.x) | ((u32)f2bf(v.y) << 16);
  o[1] = (u32)f2bf(v.z) | ((u32)f2bf(v.w) << 16);
  *(u32x2*)(out + i) = o;
}

// ---------------- QKV projection (+RoPE) ----------------
// MODE 0: Q -> rope, write (NH,S,D);  MODE 1: K -> rope, write (NKV,S,D)
// MODE 2: V -> no rope, write transposed (NKV,D,S)
template <int MODE>
__global__ __launch_bounds__(128) void proj_kernel(const u16* __restrict__ X,   // (S,H) bf16
                                                   const u16* __restrict__ W,   // (O,H) bf16
                                                   const float* __restrict__ cosp,
                                                   const float* __restrict__ sinp,
                                                   u16* __restrict__ dst) {
  int lane = threadIdx.x & 31;
  int wave = threadIdx.x >> 5;
  int half = lane >> 4;
  int lm   = lane & 15;
  int s0   = blockIdx.x * 64 + wave * 16;
  int head = blockIdx.y;
  int n0   = head * 128;

  v8f acc[8];
#pragma unroll
  for (int t = 0; t < 8; ++t) acc[t] = vzero();

  const u16* xb = X + (size_t)s0 * H_;
  for (int k = 0; k < H_; k += 32) {
    v16bf a = load_frag_a(xb + k, H_, lane);
#pragma unroll
    for (int t = 0; t < 8; ++t) {
      v16bf b = load_frag_b(W + (size_t)(n0 + 16 * t) * H_ + k, H_, lane);
      acc[t] = wmma_bf16(a, b, acc[t]);
    }
  }

  if (MODE == 2) {
    // transposed store: dst[(head*D + d)*S + s]; per lane: 8 consecutive s -> one B128
#pragma unroll
    for (int t = 0; t < 8; ++t) {
      int d = 16 * t + lm;
      u32x4 w;
#pragma unroll
      for (int j = 0; j < 4; ++j)
        w[j] = (u32)f2bf(acc[t][2 * j]) | ((u32)f2bf(acc[t][2 * j + 1]) << 16);
      *(u32x4*)(dst + (size_t)(head * D_ + d) * S_ + s0 + 8 * half) = w;
    }
  } else {
    // RoPE: pair (d, d+64) lives in accumulators (t, t+4) at identical lane/vgpr slot
#pragma unroll
    for (int t = 0; t < 4; ++t) {
      int d = 16 * t + lm;
#pragma unroll
      for (int r = 0; r < 8; ++r) {
        int srow = s0 + r + 8 * half;
        float c1 = cosp[(size_t)srow * D_ + d];
        float n1 = sinp[(size_t)srow * D_ + d];
        float c2 = cosp[(size_t)srow * D_ + d + 64];
        float n2 = sinp[(size_t)srow * D_ + d + 64];
        float lo = acc[t][r];
        float hi = acc[t + 4][r];
        float o1 = lo * c1 - hi * n1;
        float o2 = hi * c2 + lo * n2;
        size_t base = ((size_t)head * S_ + srow) * D_;
        dst[base + d]      = f2bf(o1);
        dst[base + d + 64] = f2bf(o2);
      }
    }
  }
}

// ---------------- flash attention (causal, GQA) ----------------
// One 64-key block step: 4 score tiles, online softmax, 2-step PV.
template <bool MASKED>
__device__ __forceinline__ void attn_block(int k0, int q0, int lane, int half, int lm,
                                           const v16bf* qf, const u16* kbh,
                                           const u16* vbase, u16* plds,
                                           v8f* oacc, float* m_run, float* l_run) {
  const float scale = 0.08838834764831845f;  // 1/sqrt(128)
  const u16* kb = kbh + (size_t)k0 * D_;

  v8f sc[4];
#pragma unroll
  for (int j = 0; j < 4; ++j) sc[j] = vzero();
#pragma unroll
  for (int kk = 0; kk < 4; ++kk) {
#pragma unroll
    for (int j = 0; j < 4; ++j) {
      v16bf b = load_frag_b(kb + (size_t)(16 * j) * D_ + kk * 32, D_, lane);
      sc[j] = wmma_bf16(qf[kk], b, sc[j]);
    }
  }

  float alpha[8];
#pragma unroll
  for (int r = 0; r < 8; ++r) {
    int qa = q0 + r + 8 * half;
    float x[4];
#pragma unroll
    for (int j = 0; j < 4; ++j) {
      x[j] = sc[j][r] * scale;
      if (MASKED && (k0 + 16 * j + lm > qa)) x[j] = -1e30f;
    }
    float mx = fmaxf(fmaxf(x[0], x[1]), fmaxf(x[2], x[3]));
#pragma unroll
    for (int off = 1; off < 16; off <<= 1) mx = fmaxf(mx, __shfl_xor(mx, off, 32));
    float mnew = fmaxf(m_run[r], mx);
    float rs = 0.0f;
    int m = r + 8 * half;
#pragma unroll
    for (int j = 0; j < 4; ++j) {
      float p = __expf(x[j] - mnew);
      rs += p;
      plds[m * 64 + 16 * j + lm] = f2bf(p);   // C-layout -> row-major P (16x64) in LDS
    }
#pragma unroll
    for (int off = 1; off < 16; off <<= 1) rs += __shfl_xor(rs, off, 32);
    float a = __expf(m_run[r] - mnew);
    l_run[r] = l_run[r] * a + rs;
    m_run[r] = mnew;
    alpha[r] = a;
  }

#pragma unroll
  for (int t = 0; t < 8; ++t)
#pragma unroll
    for (int r = 0; r < 8; ++r) oacc[t][r] *= alpha[r];

  v16bf pa0 = load_frag_a(plds, 64, lane);
  v16bf pa1 = load_frag_a(plds + 32, 64, lane);
#pragma unroll
  for (int t = 0; t < 8; ++t) {
    v16bf b0 = load_frag_b(vbase + (size_t)(16 * t) * S_ + k0, S_, lane);
    oacc[t] = wmma_bf16(pa0, b0, oacc[t]);
    v16bf b1 = load_frag_b(vbase + (size_t)(16 * t) * S_ + k0 + 32, S_, lane);
    oacc[t] = wmma_bf16(pa1, b1, oacc[t]);
  }
}

__global__ __launch_bounds__(128) void attn_kernel(const u16* __restrict__ Qr,  // (NH,S,D)
                                                   const u16* __restrict__ Kr,  // (NKV,S,D)
                                                   const u16* __restrict__ Vt,  // (NKV,D,S)
                                                   u16* __restrict__ Ao) {      // (S, NH*D)
  __shared__ __align__(16) u16 Plds[4][16 * 64];

  int lane = threadIdx.x & 31;
  int wave = threadIdx.x >> 5;
  int half = lane >> 4;
  int lm   = lane & 15;
  int head = blockIdx.y;
  int kvh  = head >> 2;                      // NH/NKV = 4
  int q0   = blockIdx.x * 64 + wave * 16;

  // preload Q fragments for all four 32-wide d chunks
  v16bf qf[4];
  const u16* qbase = Qr + ((size_t)head * S_ + q0) * D_;
#pragma unroll
  for (int kk = 0; kk < 4; ++kk) qf[kk] = load_frag_a(qbase + kk * 32, D_, lane);

  v8f oacc[8];
#pragma unroll
  for (int t = 0; t < 8; ++t) oacc[t] = vzero();
  float m_run[8], l_run[8];
#pragma unroll
  for (int r = 0; r < 8; ++r) { m_run[r] = -1e30f; l_run[r] = 0.0f; }

  const u16* kbh   = Kr + (size_t)kvh * S_ * D_;
  const u16* vbase = Vt + (size_t)kvh * D_ * S_;
  u16* plds = &Plds[wave][0];

  // fully-unmasked 64-key blocks: k0 + 63 <= q0 for every query row in the tile
  int nfull = q0 >> 6;
  for (int b = 0; b < nfull; ++b)
    attn_block<false>(b * 64, q0, lane, half, lm, qf, kbh, vbase, plds, oacc, m_run, l_run);
  // single diagonal (masked) block covers keys nfull*64 .. q0+15
  attn_block<true>(nfull * 64, q0, lane, half, lm, qf, kbh, vbase, plds, oacc, m_run, l_run);

  // normalize + store bf16 (S, NH*D)
#pragma unroll
  for (int r = 0; r < 8; ++r) {
    float inv = 1.0f / l_run[r];
    int srow = q0 + r + 8 * half;
#pragma unroll
    for (int t = 0; t < 8; ++t) {
      int d = 16 * t + lm;
      Ao[(size_t)srow * (NH_ * D_) + head * D_ + d] = f2bf(oacc[t][r] * inv);
    }
  }
}

// ---------------- output projection ----------------
__global__ __launch_bounds__(128) void oproj_kernel(const u16* __restrict__ A,   // (S, 2048) bf16
                                                    const u16* __restrict__ W,   // (2048, 2048) bf16
                                                    float* __restrict__ out) {   // (S, H) f32
  int lane = threadIdx.x & 31;
  int wave = threadIdx.x >> 5;
  int half = lane >> 4;
  int lm   = lane & 15;
  int s0   = blockIdx.x * 64 + wave * 16;
  int n0   = blockIdx.y * 128;

  v8f acc[8];
#pragma unroll
  for (int t = 0; t < 8; ++t) acc[t] = vzero();

  const u16* ab = A + (size_t)s0 * (NH_ * D_);
  for (int k = 0; k < NH_ * D_; k += 32) {
    v16bf a = load_frag_a(ab + k, NH_ * D_, lane);
#pragma unroll
    for (int t = 0; t < 8; ++t) {
      v16bf b = load_frag_b(W + (size_t)(n0 + 16 * t) * (NH_ * D_) + k, NH_ * D_, lane);
      acc[t] = wmma_bf16(a, b, acc[t]);
    }
  }
#pragma unroll
  for (int r = 0; r < 8; ++r) {
    int srow = s0 + r + 8 * half;
#pragma unroll
    for (int t = 0; t < 8; ++t)
      out[(size_t)srow * H_ + n0 + 16 * t + lm] = acc[t][r];
  }
}

extern "C" void kernel_launch(void* const* d_in, const int* in_sizes, int n_in,
                              void* d_out, int out_size, void* d_ws, size_t ws_size,
                              hipStream_t stream) {
  const float* hidden = (const float*)d_in[0];
  const float* cosp   = (const float*)d_in[1];
  const float* sinp   = (const float*)d_in[2];
  const float* Wq     = (const float*)d_in[3];
  const float* Wk     = (const float*)d_in[4];
  const float* Wv     = (const float*)d_in[5];
  const float* Wo     = (const float*)d_in[6];

  u16* hB  = (u16*)d_ws;                         // (S,H)
  u16* WqB = hB  + (size_t)S_ * H_;              // (2048,2048)
  u16* WkB = WqB + (size_t)NH_ * D_ * H_;        // (512,2048)
  u16* WvB = WkB + (size_t)NKV_ * D_ * H_;       // (512,2048)
  u16* WoB = WvB + (size_t)NKV_ * D_ * H_;       // (2048,2048)
  u16* Qr  = WoB + (size_t)H_ * NH_ * D_;        // (NH,S,D)
  u16* Kr  = Qr  + (size_t)NH_ * S_ * D_;        // (NKV,S,D)
  u16* Vt  = Kr  + (size_t)NKV_ * S_ * D_;       // (NKV,D,S)
  u16* AoB = Vt  + (size_t)NKV_ * S_ * D_;       // (S, NH*D)

  auto cvt = [&](const float* src, u16* dstp, int n) {
    cvt_kernel<<<dim3((n / 4 + 255) / 256), dim3(256), 0, stream>>>(src, dstp, n);
  };
  cvt(hidden, hB,  S_ * H_);
  cvt(Wq,     WqB, NH_ * D_ * H_);
  cvt(Wk,     WkB, NKV_ * D_ * H_);
  cvt(Wv,     WvB, NKV_ * D_ * H_);
  cvt(Wo,     WoB, H_ * NH_ * D_);

  proj_kernel<0><<<dim3(S_ / 64, NH_),  dim3(128), 0, stream>>>(hB, WqB, cosp, sinp, Qr);
  proj_kernel<1><<<dim3(S_ / 64, NKV_), dim3(128), 0, stream>>>(hB, WkB, cosp, sinp, Kr);
  proj_kernel<2><<<dim3(S_ / 64, NKV_), dim3(128), 0, stream>>>(hB, WvB, cosp, sinp, Vt);

  attn_kernel<<<dim3(S_ / 64, NH_), dim3(128), 0, stream>>>(Qr, Kr, Vt, AoB);

  oproj_kernel<<<dim3(S_ / 64, H_ / 128), dim3(128), 0, stream>>>(AoB, WoB, (float*)d_out);
}
